// SimpleAttention_65695819759905
// MI455X (gfx1250) — compile-verified, matrix operand order checked by reading, not run on hardware
//
#include <hip/hip_runtime.h>

#define DIMD   2048
#define NHEADS 16
#define HD     128
#define SEQ    2048
#define BATCH  4

typedef _Float16 f16;
typedef __attribute__((ext_vector_type(16))) _Float16 v16h;
typedef __attribute__((ext_vector_type(8)))  _Float16 v8h;
typedef __attribute__((ext_vector_type(8)))  float    v8f;
typedef unsigned int u32x4 __attribute__((ext_vector_type(4)));
typedef int          i32x8 __attribute__((ext_vector_type(8)));
typedef int          i32x4 __attribute__((ext_vector_type(4)));

// Load a 16x32 f16 WMMA fragment from a row-major matrix.
// Lane L holds row (row0 + L%16); K-chunks at col0+8*h and col0+8*h+16 (h = L/16).
// Serves as A (MxK row-major) or as B given as (NxK) row-major (B^T col-major).
static __device__ inline v16h load_frag(const f16* __restrict__ p, int ld,
                                        int row0, int col0, int lane) {
  int r = row0 + (lane & 15);
  int h = (lane >> 4) & 1;
  const f16* q = p + (size_t)r * (size_t)ld + col0 + 8 * h;
  v8h c0 = *(const v8h*)(q);
  v8h c1 = *(const v8h*)(q + 16);
  v16h a;
#pragma unroll
  for (int i = 0; i < 8; ++i) { a[i] = c0[i]; a[i + 8] = c1[i]; }
  return a;
}

// ---- Tensor Data Mover: DMA a 32-row x 128-half f16 tile (row stride DIMD halves)
// from global into LDS, padding each 64-DWORD row with 4 DWORDs (LDS stride 136 halves).
// D# group0/group1 per CDNA5 ISA 8.3/8.4; groups 2/3 zero (2-D tile); 6-arg builtin.
static __device__ inline void tdm_load_k_tile(unsigned lds_off, const f16* gptr) {
  unsigned long long ga = (unsigned long long)(size_t)gptr;
  u32x4 g0;
  g0[0] = 1u;                                   // count=1, user descriptor
  g0[1] = lds_off;                              // LDS byte address
  g0[2] = (unsigned)(ga & 0xFFFFFFFFu);         // global_addr[31:0]
  g0[3] = (unsigned)((ga >> 32) & 0x1FFFFFFu)   // global_addr[56:32]
          | (2u << 30);                         // type = 2 ("image")
  i32x8 g1;
  g1[0] = (1 << 16)       // data_size = 2 bytes
        | (1 << 20)       // pad_enable
        | (5 << 22)       // pad_interval: 64 DWORDs (= one 128-half row)
        | (3 << 25);      // pad_amount: 4 DWORDs (8 halves) -> row stride 136 halves
  g1[1] = (HD & 0xFFFF) << 16;                  // tensor_dim0[15:0] = 128
  g1[2] = ((SEQ & 0xFFFF) << 16);               // tensor_dim0 hi | tensor_dim1[15:0] = 2048
  g1[3] = (HD << 16);                           // tensor_dim1 hi | tile_dim0 = 128
  g1[4] = 32;                                   // tile_dim1 = 32, tile_dim2 = 0 (unused)
  g1[5] = DIMD;                                 // tensor_dim0_stride = 2048
  g1[6] = 0;
  g1[7] = 0;
  i32x4 gz4 = {0, 0, 0, 0};
  i32x8 gz8 = {0, 0, 0, 0, 0, 0, 0, 0};
  __builtin_amdgcn_tensor_load_to_lds(g0, g1, gz4, gz4, gz8, 0);
}

// ---------------- elementwise casts ----------------
__global__ void cast_f32_to_f16(const float* __restrict__ in, f16* __restrict__ out, size_t n) {
  size_t i = (size_t)blockIdx.x * blockDim.x + threadIdx.x;
  if (i < n) out[i] = (f16)in[i];
}

// ---------------- RoPE (in place on f16, f32 math), optional scale fold ----------------
__global__ void rope_inplace(f16* __restrict__ T, float scale, int total) {
  int i = blockIdx.x * blockDim.x + threadIdx.x;
  if (i >= total) return;
  int d  = i & 63;          // hd/2 = 64
  int t  = i >> 6;
  int hh = t & (NHEADS - 1);
  t >>= 4;
  int s  = t & (SEQ - 1);
  int bb = t >> 11;         // SEQ = 2^11
  size_t base = ((size_t)(bb * SEQ + s)) * DIMD + hh * HD + 2 * d;
  float t1 = (float)T[base];
  float t2 = (float)T[base + 1];
  float ang = (float)s * powf(10000.0f, -(float)d * (1.0f / 64.0f));
  float sn, cs;
  sincosf(ang, &sn, &cs);
  T[base]     = (f16)((t1 * cs - t2 * sn) * scale);
  T[base + 1] = (f16)((t1 * sn + t2 * cs) * scale);
}

// ---------------- WMMA GEMM: C[M,N] = A[M,K] * B[N,K]^T ----------------
// block = 256 threads = 8 waves; each wave -> 16(M) x 64(N); grid (M/128, N/64)
template <typename OutT>
__global__ void __launch_bounds__(256)
gemm_xWT(const f16* __restrict__ A, const f16* __restrict__ B,
         OutT* __restrict__ C, int M, int N, int K) {
  int lane = threadIdx.x & 31;
  int w    = threadIdx.x >> 5;
  int m0   = (blockIdx.x * 8 + w) * 16;
  int n0   = blockIdx.y * 64;

  v8f zf = {0.f, 0.f, 0.f, 0.f, 0.f, 0.f, 0.f, 0.f};
  v8f acc[4];
#pragma unroll
  for (int j = 0; j < 4; ++j) acc[j] = zf;

  for (int k0 = 0; k0 < K; k0 += 32) {
    v16h a = load_frag(A, K, m0, k0, lane);
#pragma unroll
    for (int j = 0; j < 4; ++j) {
      v16h b = load_frag(B, K, n0 + 16 * j, k0, lane);
      acc[j] = __builtin_amdgcn_wmma_f32_16x16x32_f16(false, a, false, b,
                                                      (short)0, acc[j], false, false);
    }
  }
  int g = lane >> 4, n = lane & 15;
#pragma unroll
  for (int j = 0; j < 4; ++j)
#pragma unroll
    for (int v = 0; v < 8; ++v)
      C[(size_t)(m0 + v + 8 * g) * (size_t)N + n0 + 16 * j + n] = (OutT)acc[j][v];
}

// ---------------- Flash attention (transposed scores, double-buffered TDM) ----------
// grid (SEQ/64, NHEADS, BATCH), block 128 (4 waves). Wave w owns 16 queries.
// S^T = K_tile x Q^T  -> lane column = query; softmax in-lane + one xor(16).
// P^T stays in registers as the B fragment of O^T = V^T x P.
__global__ void __launch_bounds__(128)
flash_attn(const f16* __restrict__ Qh, const f16* __restrict__ Kh,
           const f16* __restrict__ Vh, f16* __restrict__ Oh) {
  __shared__ f16 Klds[2][32 * 136]; // double-buffered K tile, TDM-padded stride 136 halves
  __shared__ f16 Vt[HD * 40];       // V tile transposed: (d, k), row stride 40 halves

  int lane = threadIdx.x & 31;
  int w    = threadIdx.x >> 5;
  int q0   = blockIdx.x * 64 + w * 16;
  int h    = blockIdx.y;
  int b    = blockIdx.z;
  int col  = h * HD;

  const f16* Qb = Qh + (size_t)b * SEQ * DIMD;
  const f16* Kb = Kh + (size_t)b * SEQ * DIMD;
  const f16* Vb = Vh + (size_t)b * SEQ * DIMD;
  unsigned klds_off0 = (unsigned)(size_t)(void*)&Klds[0][0];  // low 32 bits = LDS offset
  unsigned klds_off1 = (unsigned)(size_t)(void*)&Klds[1][0];

  // persistent Q fragments (WMMA B operand): 16 queries x 128 d, 4 K-steps
  v16h qf[4];
#pragma unroll
  for (int ks = 0; ks < 4; ++ks) qf[ks] = load_frag(Qb, DIMD, q0, col + 32 * ks, lane);

  v8f zf = {0.f, 0.f, 0.f, 0.f, 0.f, 0.f, 0.f, 0.f};
  v8f o[8];                 // O^T accumulators: o[dt] rows d = dt*16 + v + 8g, col = query
#pragma unroll
  for (int dt = 0; dt < 8; ++dt) o[dt] = zf;
  float rmax = -1e30f, rsum = 0.0f;   // per-lane stats of this lane's query column

  int tk  = threadIdx.x & 31;  // V staging: key row within tile
  int sub = threadIdx.x >> 5;  // V staging: which 32-wide d slice

  // prologue: DMA first K tile into buffer 0
  if (w == 0) tdm_load_k_tile(klds_off0, Kb + col);

  for (int kt = 0; kt < SEQ; kt += 32) {
    int buf = (kt >> 5) & 1;
    const f16* Kp = &Klds[buf][0];

    // ---- issue next tile's DMA into the other buffer (hidden behind this tile) ----
    if (w == 0 && (kt + 32) < SEQ)
      tdm_load_k_tile(buf ? klds_off0 : klds_off1, Kb + (size_t)(kt + 32) * DIMD + col);

    // ---- stage V tile transposed into LDS (all 4 waves) ----
#pragma unroll
    for (int c = 0; c < 4; ++c) {
      v8h vv = *(const v8h*)(Vb + (size_t)(kt + tk) * DIMD + col + sub * 32 + c * 8);
#pragma unroll
      for (int i = 0; i < 8; ++i) Vt[(sub * 32 + c * 8 + i) * 40 + tk] = vv[i];
    }
    // TDM ops from one wave are in-order: <=1 outstanding means current tile landed.
    if (w == 0) {
      if ((kt + 32) < SEQ) __builtin_amdgcn_s_wait_tensorcnt(1);
      else                 __builtin_amdgcn_s_wait_tensorcnt(0);
    }
    __syncthreads();

    // ---- transposed scores: S^T = K_tile (32x128) x Q^T, pipelined fragment loads ----
    v8f s0 = zf, s1 = zf;   // s0: keys kt+{v+8g}; s1: keys kt+16+{v+8g}
    v16h ka = load_frag(Kp, 136, 0,  0, lane);
    v16h kb = load_frag(Kp, 136, 16, 0, lane);
#pragma unroll
    for (int ks = 0; ks < 4; ++ks) {
      v16h na = ka, nb = kb;
      if (ks < 3) {
        ka = load_frag(Kp, 136, 0,  32 * (ks + 1), lane);
        kb = load_frag(Kp, 136, 16, 32 * (ks + 1), lane);
      }
      s0 = __builtin_amdgcn_wmma_f32_16x16x32_f16(false, na, false, qf[ks], (short)0, s0, false, false);
      s1 = __builtin_amdgcn_wmma_f32_16x16x32_f16(false, nb, false, qf[ks], (short)0, s1, false, false);
    }

    // ---- online softmax over keys: in-lane over 16 regs + one xor(16) ----
    float mx = fmaxf(s0[0], s1[0]);
#pragma unroll
    for (int v = 1; v < 8; ++v) mx = fmaxf(mx, fmaxf(s0[v], s1[v]));
    mx = fmaxf(mx, __shfl_xor(mx, 16));
    float nm   = fmaxf(rmax, mx);
    float corr = __expf(rmax - nm);
    rmax = nm;

    v16h pv;                 // P^T fragment: e<8 -> key e+8g ; e>=8 -> key (e-8)+8g+16
    float psum = 0.0f;
#pragma unroll
    for (int v = 0; v < 8; ++v) {
      float p0 = __expf(s0[v] - nm);
      float p1 = __expf(s1[v] - nm);
      psum += p0 + p1;
      pv[v]     = (f16)p0;
      pv[v + 8] = (f16)p1;
    }
    psum += __shfl_xor(psum, 16);
    rsum = rsum * corr + psum;

    // ---- O^T = corr*O^T + V^T (16d x 32k) x P (32k x 16q), pipelined A loads ----
    v16h va = load_frag(Vt, 40, 0, 0, lane);
#pragma unroll
    for (int dt = 0; dt < 8; ++dt) {
      v16h cur = va;
      if (dt < 7) va = load_frag(Vt, 40, (dt + 1) * 16, 0, lane);
      v8f oc = o[dt] * corr;
      o[dt] = __builtin_amdgcn_wmma_f32_16x16x32_f16(false, cur, false, pv, (short)0, oc, false, false);
    }
    __syncthreads();   // protect Klds/Vt before next tile's DMA/staging
  }

  // ---- normalize and store (contiguous v8h per (dt, g)) ----
  int g = lane >> 4, n = lane & 15;
  float inv = 1.0f / rsum;
  size_t rowbase = ((size_t)b * SEQ + q0 + n) * DIMD + col;
#pragma unroll
  for (int dt = 0; dt < 8; ++dt) {
    v8h ov;
#pragma unroll
    for (int v = 0; v < 8; ++v) ov[v] = (f16)(o[dt][v] * inv);
    *(v8h*)(Oh + rowbase + dt * 16 + 8 * g) = ov;
  }
}

// ---------------- launcher ----------------
extern "C" void kernel_launch(void* const* d_in, const int* in_sizes, int n_in,
                              void* d_out, int out_size, void* d_ws, size_t ws_size,
                              hipStream_t stream) {
  const float* x  = (const float*)d_in[0];
  const float* Wq = (const float*)d_in[1];
  const float* Wk = (const float*)d_in[2];
  const float* Wv = (const float*)d_in[3];
  const float* Wo = (const float*)d_in[4];
  float* out = (float*)d_out;

  const size_t MT = (size_t)BATCH * SEQ;   // 8192 rows
  const size_t NX = MT * DIMD;             // activation elements
  const size_t NW = (size_t)DIMD * DIMD;   // weight elements

  char* ws = (char*)d_ws;
  f16* xh  = (f16*)ws; ws += NX * 2;
  f16* Wqh = (f16*)ws; ws += NW * 2;
  f16* Wkh = (f16*)ws; ws += NW * 2;
  f16* Wvh = (f16*)ws; ws += NW * 2;
  f16* Woh = (f16*)ws; ws += NW * 2;
  f16* Qh  = (f16*)ws; ws += NX * 2;
  f16* Kh  = (f16*)ws; ws += NX * 2;
  f16* Vh  = (f16*)ws; ws += NX * 2;
  f16* Oh  = (f16*)ws; ws += NX * 2;

  // 1) casts to f16
  cast_f32_to_f16<<<(unsigned)((NX + 255) / 256), 256, 0, stream>>>(x,  xh,  NX);
  cast_f32_to_f16<<<(unsigned)((NW + 255) / 256), 256, 0, stream>>>(Wq, Wqh, NW);
  cast_f32_to_f16<<<(unsigned)((NW + 255) / 256), 256, 0, stream>>>(Wk, Wkh, NW);
  cast_f32_to_f16<<<(unsigned)((NW + 255) / 256), 256, 0, stream>>>(Wv, Wvh, NW);
  cast_f32_to_f16<<<(unsigned)((NW + 255) / 256), 256, 0, stream>>>(Wo, Woh, NW);

  // 2) QKV projections (f16 out, f32 accumulate inside WMMA)
  dim3 ggrid((unsigned)(MT / 128), DIMD / 64);
  gemm_xWT<f16><<<ggrid, 256, 0, stream>>>(xh, Wqh, Qh, (int)MT, DIMD, DIMD);
  gemm_xWT<f16><<<ggrid, 256, 0, stream>>>(xh, Wkh, Kh, (int)MT, DIMD, DIMD);
  gemm_xWT<f16><<<ggrid, 256, 0, stream>>>(xh, Wvh, Vh, (int)MT, DIMD, DIMD);

  // 3) RoPE in place; fold softmax scale 1/sqrt(hd) into Q
  int pairs = (int)(NX / 2);
  rope_inplace<<<(pairs + 255) / 256, 256, 0, stream>>>(Qh, 0.08838834764831845f, pairs);
  rope_inplace<<<(pairs + 255) / 256, 256, 0, stream>>>(Kh, 1.0f, pairs);

  // 4) flash attention (double-buffered TDM K tiles, register-resident P)
  flash_attn<<<dim3(SEQ / 64, NHEADS, BATCH), 128, 0, stream>>>(Qh, Kh, Vh, Oh);

  // 5) output projection (f32 out)
  gemm_xWT<float><<<ggrid, 256, 0, stream>>>(Oh, Woh, out, (int)MT, DIMD, DIMD);
}
